// MoEGate_721554506201
// MI455X (gfx1250) — compile-verified, matrix operand order checked by reading, not run on hardware
//
#include <hip/hip_runtime.h>

typedef __attribute__((ext_vector_type(16))) __bf16 v16bf;
typedef __attribute__((ext_vector_type(8)))  __bf16 v8bf;
typedef __attribute__((ext_vector_type(8)))  float  v8f;
typedef __attribute__((ext_vector_type(4)))  float  v4f;

#define T_TOK   32768
#define B_DIM   8
#define S_DIM   4096
#define H_DIM   1024
#define E_DIM   64
#define TOPK    2
#define ALPHA_F 0.1f
#define MTILES  2            // token tiles (of 16) per wave
#define WAVES   8            // waves per block
#define TOK_PER_BLOCK (MTILES * 16 * WAVES)   // 256

// ---------------- weight -> bf16 hi/lo split (runs once per launch) ----------
__global__ void convert_w_kernel(const float* __restrict__ w,
                                 __bf16* __restrict__ w_hi,
                                 __bf16* __restrict__ w_lo) {
    int i = blockIdx.x * blockDim.x + threadIdx.x;     // grid covers E*H exactly
    float f = w[i];
    __bf16 h = (__bf16)f;
    w_hi[i] = h;
    w_lo[i] = (__bf16)(f - (float)h);
}

// ---------------- main router kernel ----------------------------------------
__global__ __launch_bounds__(256) void moe_gate_kernel(
        const float*  __restrict__ x,
        const __bf16* __restrict__ w_hi,
        const __bf16* __restrict__ w_lo,
        float* __restrict__ out_idx,   // [T][2] (indices stored as float)
        float* __restrict__ out_w,     // [T][2]
        float* __restrict__ g_sum,     // [B][E] global score sums
        float* __restrict__ g_cnt)     // [B][E] global top-k counts
{
    __shared__ float lds_logits[WAVES * 16 * E_DIM];   // 32 KB: one 16-tok pass
    __shared__ float lds_sum[E_DIM];
    __shared__ float lds_cnt[E_DIM];

    const int tid  = threadIdx.x;
    const int wave = tid >> 5;
    const int lane = tid & 31;

    if (tid < E_DIM) { lds_sum[tid] = 0.0f; lds_cnt[tid] = 0.0f; }
    __syncthreads();

    const int blockTok = blockIdx.x * TOK_PER_BLOCK;
    const int tokBase  = blockTok + wave * (MTILES * 16);
    const int row      = lane & 15;          // token-in-tile (A) / expert-in-tile (B)
    const int kOff     = (lane >> 4) * 8;    // ISA 16-bit A/B lane K striping

    const float* xrow[MTILES];
#pragma unroll
    for (int mt = 0; mt < MTILES; ++mt)
        xrow[mt] = x + (size_t)(tokBase + mt * 16 + row) * H_DIM;

    v8f acc[MTILES][4];
    const v8f vzero = {0.f,0.f,0.f,0.f,0.f,0.f,0.f,0.f};
#pragma unroll
    for (int mt = 0; mt < MTILES; ++mt)
#pragma unroll
        for (int t = 0; t < 4; ++t) acc[mt][t] = vzero;

    for (int k0 = 0; k0 < H_DIM; k0 += 32) {
        // ---- A fragments (fp32 -> bf16 hi/lo), layout per ISA 16-bit A 16x32
        v16bf a_hi[MTILES], a_lo[MTILES];
#pragma unroll
        for (int mt = 0; mt < MTILES; ++mt) {
            __builtin_prefetch(xrow[mt] + k0 + 32, 0, 0);
            v4f f0 = *(const v4f*)(xrow[mt] + k0 + kOff);
            v4f f1 = *(const v4f*)(xrow[mt] + k0 + kOff + 4);
            v4f f2 = *(const v4f*)(xrow[mt] + k0 + kOff + 16);
            v4f f3 = *(const v4f*)(xrow[mt] + k0 + kOff + 20);
#pragma unroll
            for (int i = 0; i < 4; ++i) {
                float v; __bf16 h;
                v = f0[i]; h = (__bf16)v; a_hi[mt][i]      = h; a_lo[mt][i]      = (__bf16)(v - (float)h);
                v = f1[i]; h = (__bf16)v; a_hi[mt][4 + i]  = h; a_lo[mt][4 + i]  = (__bf16)(v - (float)h);
                v = f2[i]; h = (__bf16)v; a_hi[mt][8 + i]  = h; a_lo[mt][8 + i]  = (__bf16)(v - (float)h);
                v = f3[i]; h = (__bf16)v; a_hi[mt][12 + i] = h; a_lo[mt][12 + i] = (__bf16)(v - (float)h);
            }
        }
        // ---- 4 expert tiles of 16; B rows of weight are exactly the B striping
#pragma unroll
        for (int t = 0; t < 4; ++t) {
            const size_t eoff = (size_t)(t * 16 + row) * H_DIM + k0 + kOff;
            v8bf bh0 = *(const v8bf*)(w_hi + eoff);
            v8bf bh1 = *(const v8bf*)(w_hi + eoff + 16);
            v8bf bl0 = *(const v8bf*)(w_lo + eoff);
            v8bf bl1 = *(const v8bf*)(w_lo + eoff + 16);
            v16bf b_hi = __builtin_shufflevector(bh0, bh1, 0,1,2,3,4,5,6,7,8,9,10,11,12,13,14,15);
            v16bf b_lo = __builtin_shufflevector(bl0, bl1, 0,1,2,3,4,5,6,7,8,9,10,11,12,13,14,15);
#pragma unroll
            for (int mt = 0; mt < MTILES; ++mt) {
                acc[mt][t] = __builtin_amdgcn_wmma_f32_16x16x32_bf16(
                    false, a_hi[mt], false, b_hi, (short)0, acc[mt][t], false, false);
                acc[mt][t] = __builtin_amdgcn_wmma_f32_16x16x32_bf16(
                    false, a_hi[mt], false, b_lo, (short)0, acc[mt][t], false, false);
                acc[mt][t] = __builtin_amdgcn_wmma_f32_16x16x32_bf16(
                    false, a_lo[mt], false, b_hi, (short)0, acc[mt][t], false, false);
            }
        }
    }

    // ---- epilogue: two passes of 16 tokens/wave through 32 KB LDS -----------
    const int halfM = (lane >> 4) * 8;       // C/D layout: lanes 16-31 hold M=8..15
    float* wlds = lds_logits + wave * (16 * E_DIM);

    for (int mt = 0; mt < MTILES; ++mt) {
#pragma unroll
        for (int r = 0; r < 8; ++r) {
            const int m = r + halfM;
#pragma unroll
            for (int t = 0; t < 4; ++t)
                wlds[m * E_DIM + t * 16 + row] = acc[mt][t][r];
        }
        __syncthreads();

        if (lane < 16) {
            const float* lrow = wlds + lane * E_DIM;
            float mx = lrow[0];
            for (int e = 1; e < E_DIM; ++e) mx = fmaxf(mx, lrow[e]);
            float Z = 0.0f;
            for (int e = 0; e < E_DIM; ++e) Z += __expf(lrow[e] - mx);
            const float inv = 1.0f / Z;
            float v1 = -1.0f, v2 = -1.0f; int i1 = 0, i2 = 0;
            for (int e = 0; e < E_DIM; ++e) {
                float s = __expf(lrow[e] - mx) * inv;       // softmax score
                atomicAdd(&lds_sum[e], s);                  // for mean_scores
                if (s > v1)      { v2 = v1; i2 = i1; v1 = s; i1 = e; }
                else if (s > v2) { v2 = s;  i2 = e; }
            }
            atomicAdd(&lds_cnt[i1], 1.0f);
            atomicAdd(&lds_cnt[i2], 1.0f);
            const int token = tokBase + mt * 16 + lane;
            const float wn  = 1.0f / (v1 + v2 + 1e-20f);
            out_idx[token * 2 + 0] = (float)i1;
            out_idx[token * 2 + 1] = (float)i2;
            out_w  [token * 2 + 0] = v1 * wn;
            out_w  [token * 2 + 1] = v2 * wn;
        }
        __syncthreads();
    }

    if (tid < E_DIM) {       // one global atomic per (block, expert)
        const int b = blockTok / S_DIM;      // blocks never straddle a batch
        atomicAdd(&g_sum[b * E_DIM + tid], lds_sum[tid]);
        atomicAdd(&g_cnt[b * E_DIM + tid], lds_cnt[tid]);
    }
}

// ---------------- aux-loss reduction -----------------------------------------
__global__ void aux_kernel(const float* __restrict__ g_sum,
                           const float* __restrict__ g_cnt,
                           float* __restrict__ aux_out) {
    __shared__ float red[E_DIM];
    const int e = threadIdx.x;
    float p = 0.0f;
    for (int b = 0; b < B_DIM; ++b)
        p += g_cnt[b * E_DIM + e] * g_sum[b * E_DIM + e];
    red[e] = p;
    __syncthreads();
    if (e == 0) {
        float s = 0.0f;
        for (int i = 0; i < E_DIM; ++i) s += red[i];
        // aux = alpha * mean_b sum_e (cnt*E/(S*K)) * (sum/S)
        const float scale = ALPHA_F * (float)E_DIM /
            ((float)B_DIM * (float)S_DIM * (float)TOPK * (float)S_DIM);
        *aux_out = s * scale;
    }
}

extern "C" void kernel_launch(void* const* d_in, const int* in_sizes, int n_in,
                              void* d_out, int out_size, void* d_ws, size_t ws_size,
                              hipStream_t stream) {
    (void)in_sizes; (void)n_in; (void)out_size; (void)ws_size;
    const float* x = (const float*)d_in[0];     // [B,S,H] fp32
    const float* w = (const float*)d_in[1];     // [E,H]   fp32
    float* out     = (float*)d_out;             // idx[T*2] | weights[T*2] | aux

    // workspace layout: [B*E sums][B*E counts] pad to 4KB | w_hi bf16 | w_lo bf16
    float*  g_sum = (float*)d_ws;
    float*  g_cnt = g_sum + B_DIM * E_DIM;
    __bf16* w_hi  = (__bf16*)((char*)d_ws + 4096);
    __bf16* w_lo  = (__bf16*)((char*)d_ws + 4096 + (size_t)E_DIM * H_DIM * sizeof(__bf16));

    hipMemsetAsync(d_ws, 0, 2 * B_DIM * E_DIM * sizeof(float), stream);
    convert_w_kernel<<<(E_DIM * H_DIM) / 256, 256, 0, stream>>>(w, w_hi, w_lo);
    moe_gate_kernel<<<T_TOK / TOK_PER_BLOCK, 256, 0, stream>>>(
        x, w_hi, w_lo, out, out + T_TOK * 2, g_sum, g_cnt);
    aux_kernel<<<1, E_DIM, 0, stream>>>(g_sum, g_cnt, out + T_TOK * 4);
}
